// Attention_1271310320050
// MI455X (gfx1250) — matrix operand order repaired and checked by reading, not compiled, not run on hardware
//
#include <hip/hip_runtime.h>

typedef unsigned short u16;
typedef unsigned int   u32;
typedef unsigned long long u64;
typedef __attribute__((ext_vector_type(16))) __bf16 v16bf;
typedef __attribute__((ext_vector_type(8)))  float  v8f;
typedef __attribute__((ext_vector_type(4)))  u32    u32x4;
typedef __attribute__((ext_vector_type(8)))  int    i32x8;
typedef __attribute__((ext_vector_type(4)))  int    i32x4;

#define B_   8
#define N_   1024
#define C_   1024
#define H_   16
#define D_   64
#define C3_  3072
#define SCALE_ 0.125f   // 64^-0.5

union FragU { v16bf v; u32 u[8]; };

__device__ __forceinline__ u16 f2bf(float f) {
  u32 x = __float_as_uint(f);
  u32 r = x + 0x7FFFu + ((x >> 16) & 1u);
  return (u16)(r >> 16);
}

__device__ __forceinline__ v8f wmma_bf16(v16bf a, v16bf b, v8f c) {
  return __builtin_amdgcn_wmma_f32_16x16x32_bf16(false, a, false, b, (short)0, c,
                                                 false, false);
}

// A fragment: 16x32 bf16 tile (contiguous k order: k = 16*(lane/16) + s).
__device__ __forceinline__ v16bf load_A_lds(const u16* base, int stride, int m0,
                                            int k0, int lane) {
  FragU f;
  const u16* p = base + (m0 + (lane & 15)) * stride + k0;
  int h8 = (lane >> 4) << 4;
#pragma unroll
  for (int j = 0; j < 4; ++j) {
    f.u[j]     = *(const u32*)(p + h8 + 2 * j);
    f.u[j + 4] = *(const u32*)(p + 8 + h8 + 2 * j);
  }
  return f.v;
}

// B fragment: 32x16, LDS staged K-major: B[k][col] at baseT[col*strideK + k].
__device__ __forceinline__ v16bf load_Bt_lds(const u16* baseT, int strideK,
                                             int n0, int k0, int lane) {
  FragU f;
  const u16* p = baseT + (n0 + (lane & 15)) * strideK + k0 + ((lane >> 4) << 4);
#pragma unroll
  for (int j = 0; j < 8; ++j) f.u[j] = *(const u32*)(p + 2 * j);
  return f.v;
}

// ---- Tensor Data Mover: 2-D bf16 tile (width x height) global->LDS ----
// D# per CDNA5 ISA 8.3/8.4: group0 = {count|flags, lds_addr, global_addr, type=2},
// group1 = {data_size=2B, tensor_dim0/1, tile_dim0/1, dim0_stride}.
// Toolchain uses the 6-arg builtin: (g0, g1, g2, g3, g_extra, cpol).
__device__ __forceinline__ void tdm_load_2d(u32 lds_addr, const void* gaddr,
                                            u32 width_elts, u32 height,
                                            u32 row_stride_elts) {
  u64 ga = (u64)gaddr;
  u32x4 g0;
  g0[0] = 1u;                               // count=1 (valid user descriptor)
  g0[1] = lds_addr;                         // LDS byte address
  g0[2] = (u32)ga;                          // global_addr[31:0]
  g0[3] = (u32)(ga >> 32) | (2u << 30);     // global_addr[56:32] | type=2
  i32x8 g1;
  g1[0] = (int)(1u << 16);                  // data_size=1 (2 bytes/elt)
  g1[1] = (int)(width_elts << 16);          // tensor_dim0[15:0] @ bits 63:48
  g1[2] = (int)(height << 16);              // tensor_dim1[15:0] @ bits 95:80
  g1[3] = (int)(width_elts << 16);          // tile_dim0 @ bits 127:112
  g1[4] = (int)height;                      // tile_dim1 @ bits 143:128
  g1[5] = (int)row_stride_elts;             // tensor_dim0_stride[31:0]
  g1[6] = 0;
  g1[7] = 0;
  i32x4 z4 = {0, 0, 0, 0};
  i32x8 z8 = {0, 0, 0, 0, 0, 0, 0, 0};
  __builtin_amdgcn_tensor_load_to_lds(g0, g1, z4, z4, z8, 0);
}

__device__ __forceinline__ u32 lds_off(const void* p) {
  return (u32)(u64)p;   // generic LDS address: low 32 bits == LDS byte offset
}

// ---------------------------------------------------------------- convert
__global__ void cvt_f32_to_bf16(const float* __restrict__ in,
                                u16* __restrict__ out, int n4) {
  int i = blockIdx.x * blockDim.x + threadIdx.x;
  if (i < n4) {
    float4 v = ((const float4*)in)[i];
    ushort4 o;
    o.x = f2bf(v.x); o.y = f2bf(v.y); o.z = f2bf(v.z); o.w = f2bf(v.w);
    ((ushort4*)out)[i] = o;
  }
}

// ---------------------------------------------------------------- GEMM
// C[M,N] = A[M,K](bf16) * Bw[K,N](bf16); block tile 128x64, wave tile 32x32.
#define GBM 128
#define GBN 64
#define GBK 32

__global__ __launch_bounds__(256) void gemm_bf16(
    const u16* __restrict__ A, const u16* __restrict__ Bw, u16* __restrict__ Cb,
    float* __restrict__ Cf, const float* __restrict__ bias, int M, int N, int K) {
  __shared__ u16 As[GBM * GBK];   // [row][k] 8 KB, filled by TDM
  __shared__ u16 Bt[GBN * GBK];   // [col][k] 4 KB, manual transposed stage
  int t = threadIdx.x;
  int lane = t & 31, w = t >> 5;
  int m0 = blockIdx.y * GBM;
  int n0 = blockIdx.x * GBN;
  int wm = (w & 3) * 32;          // wave M offset
  int wn = (w >> 2) * 32;         // wave N offset
  v8f acc00 = {}, acc01 = {}, acc10 = {}, acc11 = {};

  int bi = t * 8;
  int br = bi >> 6, bc = bi & 63; // B stage: 8 contiguous along N -> scatter
  u32 as_lds = lds_off(As);

  for (int k0 = 0; k0 < K; k0 += GBK) {
    __syncthreads();
    uint4 bv = *(const uint4*)(Bw + (size_t)(k0 + br) * N + n0 + bc);
    const u16* bvp = (const u16*)&bv;
#pragma unroll
    for (int i = 0; i < 8; ++i) Bt[(bc + i) * GBK + br] = bvp[i];
    if (w == 0) {                 // DMA the 128x32 A tile
      tdm_load_2d(as_lds, A + (size_t)m0 * K + k0, GBK, GBM, K);
      __builtin_amdgcn_s_wait_tensorcnt(0);
    }
    if (k0 + GBK < K)
      __builtin_prefetch(Bw + (size_t)(k0 + GBK + br) * N + n0 + bc, 0, 1);
    __syncthreads();
    v16bf a0 = load_A_lds(As, GBK, wm, 0, lane);
    v16bf a1 = load_A_lds(As, GBK, wm + 16, 0, lane);
    v16bf b0 = load_Bt_lds(Bt, GBK, wn, 0, lane);
    v16bf b1 = load_Bt_lds(Bt, GBK, wn + 16, 0, lane);
    acc00 = wmma_bf16(a0, b0, acc00);
    acc01 = wmma_bf16(a0, b1, acc01);
    acc10 = wmma_bf16(a1, b0, acc10);
    acc11 = wmma_bf16(a1, b1, acc11);
  }

  int colL = lane & 15;
  int rhalf = (lane >> 4) << 3;
#pragma unroll
  for (int i = 0; i < 2; ++i) {
#pragma unroll
    for (int j = 0; j < 2; ++j) {
      v8f acc = (i == 0) ? (j == 0 ? acc00 : acc01) : (j == 0 ? acc10 : acc11);
      int col = n0 + wn + j * 16 + colL;
      float bvs = bias ? bias[col] : 0.0f;
#pragma unroll
      for (int r = 0; r < 8; ++r) {
        int row = m0 + wm + i * 16 + rhalf + r;
        float val = acc[r] + bvs;
        if (Cf) Cf[(size_t)row * N + col] = val;
        else    Cb[(size_t)row * N + col] = f2bf(val);
      }
    }
  }
}

// ---------------------------------------------------------------- attention
#define ABM 32

__global__ __launch_bounds__(256) void attn_dual(
    const u16* __restrict__ Qb, const u16* __restrict__ Kb,
    const u16* __restrict__ Vb, u16* __restrict__ Ob) {
  extern __shared__ char smem[];
  float* S  = (float*)smem;                       // 32 x 1024 f32  (128 KB)
  u16*  P   = (u16*)(smem + ABM * N_ * 4);        // 32 x 1024 bf16 ( 64 KB)
  u16*  Qs  = P + ABM * N_;                       // 32 x 64        (  4 KB)
  u16*  KVs = Qs + ABM * D_;                      // 64 x 64        (  8 KB)

  int t = threadIdx.x, lane = t & 31, w = t >> 5;
  int rb = blockIdx.x, h = blockIdx.y, b = blockIdx.z;
  size_t rowBase = (size_t)b * N_ * C3_;
  int hoff = h * D_;
  u32 qs_lds = lds_off(Qs);
  u32 kvs_lds = lds_off(KVs);

  // DMA the Q block (32 rows x 64 head cols, row stride 3C)
  if (w == 0) {
    tdm_load_2d(qs_lds, Qb + rowBase + (size_t)(rb * ABM) * C3_ + hoff,
                D_, ABM, C3_);
    __builtin_amdgcn_s_wait_tensorcnt(0);
  }

  int mt = (w >> 2) * 16;   // 0/16
  int nt = (w & 3) * 16;    // 0..48

  // ---- phase 1: S = scale * Q K^T ----
  for (int nc = 0; nc < N_; nc += 64) {
    __syncthreads();
    if (w == 0) {           // DMA K tile 64 x 64 (row-major == K-major for B)
      tdm_load_2d(kvs_lds, Kb + rowBase + (size_t)nc * C3_ + hoff, D_, 64, C3_);
      __builtin_amdgcn_s_wait_tensorcnt(0);
    }
    __syncthreads();
    v8f acc = {};
#pragma unroll
    for (int kd = 0; kd < D_; kd += 32) {
      v16bf a  = load_A_lds(Qs, D_, mt, kd, lane);
      v16bf bb = load_Bt_lds(KVs, D_, nt, kd, lane);
      acc = wmma_bf16(a, bb, acc);
    }
    int row = mt + ((lane >> 4) << 3);
    int col = nc + nt + (lane & 15);
#pragma unroll
    for (int r = 0; r < 8; ++r) S[(row + r) * N_ + col] = acc[r] * SCALE_;
  }
  __syncthreads();

  // ---- phase 2: row softmax (wave32 shuffle reductions) ----
  for (int rr = 0; rr < 4; ++rr) {
    int row = w * 4 + rr;
    float* Srow = S + row * N_;
    float mx = -3.4e38f;
    for (int c = lane; c < N_; c += 32) mx = fmaxf(mx, Srow[c]);
#pragma unroll
    for (int o = 16; o > 0; o >>= 1) mx = fmaxf(mx, __shfl_xor(mx, o, 32));
    float sum = 0.0f;
    for (int c = lane; c < N_; c += 32) {
      float e = __expf(Srow[c] - mx);
      Srow[c] = e; sum += e;
    }
#pragma unroll
    for (int o = 16; o > 0; o >>= 1) sum += __shfl_xor(sum, o, 32);
    float inv = 1.0f / sum;
    for (int c = lane; c < N_; c += 32) P[row * N_ + c] = f2bf(Srow[c] * inv);
  }
  __syncthreads();

  // ---- phase 3: O = P V (V staged transposed; TDM can't transpose) ----
  v8f acc = {};
  for (int kc = 0; kc < N_; kc += 64) {
    __syncthreads();
    { int idx = t * 8, r = idx >> 6, c = idx & 63;
      uint4 v = *(const uint4*)(Vb + rowBase + (size_t)(kc + r) * C3_ + hoff + c);
      const u16* vp = (const u16*)&v;
#pragma unroll
      for (int i = 0; i < 8; ++i) KVs[(c + i) * 64 + r] = vp[i]; }  // [d][n']
    __syncthreads();
#pragma unroll
    for (int kd = 0; kd < 64; kd += 32) {
      v16bf a  = load_A_lds(P, N_, mt, kc + kd, lane);
      v16bf bb = load_Bt_lds(KVs, 64, nt, kd, lane);
      acc = wmma_bf16(a, bb, acc);
    }
  }
  {
    int row = rb * ABM + mt + ((lane >> 4) << 3);
    int col = hoff + nt + (lane & 15);
#pragma unroll
    for (int r = 0; r < 8; ++r)
      Ob[((size_t)b * N_ + row + r) * C_ + col] = f2bf(acc[r]);
  }
}

// ---------------------------------------------------------------- launch
extern "C" void kernel_launch(void* const* d_in, const int* in_sizes, int n_in,
                              void* d_out, int out_size, void* d_ws, size_t ws_size,
                              hipStream_t stream) {
  (void)in_sizes; (void)n_in; (void)out_size; (void)ws_size;
  const float* x     = (const float*)d_in[0];
  const float* wqkv  = (const float*)d_in[1];
  const float* wproj = (const float*)d_in[2];
  const float* bproj = (const float*)d_in[3];

  u16* xb     = (u16*)d_ws;                          // 8192x1024
  u16* wqkvb  = xb     + (size_t)8192 * 1024;        // 1024x3072
  u16* wprojb = wqkvb  + (size_t)1024 * 3072;        // 1024x1024
  u16* qkv    = wprojb + (size_t)1024 * 1024;        // 8192x3072
  u16* xon    = qkv    + (size_t)8192 * 3072;        // 8192x1024 (x_new)
  u16* xoo    = xon    + (size_t)8192 * 1024;        // 8192x1024 (x_ori)

  float* outN = (float*)d_out;                       // out  (from x_new)
  float* outO = outN + (size_t)B_ * N_ * C_;         // out_ori

  cvt_f32_to_bf16<<<dim3((8192 * 1024 / 4) / 256), 256, 0, stream>>>(x, xb, 8192 * 1024 / 4);
  cvt_f32_to_bf16<<<dim3((1024 * 3072 / 4) / 256), 256, 0, stream>>>(wqkv, wqkvb, 1024 * 3072 / 4);
  cvt_f32_to_bf16<<<dim3((1024 * 1024 / 4) / 256), 256, 0, stream>>>(wproj, wprojb, 1024 * 1024 / 4);

  // QKV projection: [8192,1024] x [1024,3072] -> bf16
  gemm_bf16<<<dim3(C3_ / GBN, 8192 / GBM), 256, 0, stream>>>(
      xb, wqkvb, qkv, nullptr, nullptr, 8192, C3_, 1024);

  // dual attention
  int smem = ABM * N_ * 4 + ABM * N_ * 2 + ABM * D_ * 2 + 64 * 64 * 2; // 204 KB
  (void)hipFuncSetAttribute(reinterpret_cast<const void*>(attn_dual),
                            hipFuncAttributeMaxDynamicSharedMemorySize, smem);
  dim3 ag(N_ / ABM, H_, B_);
  attn_dual<<<ag, 256, smem, stream>>>(qkv, qkv + 1024, qkv + 2048, xoo);        // q k^T -> x_ori
  attn_dual<<<ag, 256, smem, stream>>>(qkv + 2048, qkv + 2048, qkv + 2048, xon); // v v^T -> x_new

  // output projection with bias -> f32 d_out (new first, then ori)
  gemm_bf16<<<dim3(C_ / GBN, 8192 / GBM), 256, 0, stream>>>(
      xon, wprojb, nullptr, outN, bproj, 8192, C_, 1024);
  gemm_bf16<<<dim3(C_ / GBN, 8192 / GBM), 256, 0, stream>>>(
      xoo, wprojb, nullptr, outO, bproj, 8192, C_, 1024);
}
